// model_lstm_12472585027874
// MI455X (gfx1250) — compile-verified, matrix operand order checked by reading, not run on hardware
//
#include <hip/hip_runtime.h>
#include <stdint.h>

typedef __attribute__((ext_vector_type(16))) int   v16i;
typedef __attribute__((ext_vector_type(8)))  float v8f;

#define B_TOT   2048
#define T_LEN   512
#define H_DIM   256
#define G_DIM   1024   /* 4H */
#define OUT_DIM 512
#define BTILE   16

/* ---- LDS layout (bytes) ---- */
#define OFF_W     0                      /* fp8 W_hh  [1024][256]B = 256KB  */
#define OFF_GATES (OFF_W + 262144)       /* bf16 gates [16][1024]  = 32KB   */
#define OFF_H8    (OFF_GATES + 32768)    /* fp8 h     [16][256]B   = 4KB    */
#define OFF_HF32  (OFF_H8 + 4096)        /* f32 h_last [16][256]   = 16KB   */
#define OFF_BS    (OFF_HF32 + 16384)     /* f32 (b_ih+b_hh) [1024] = 4KB    */
#define OFF_WIH   (OFF_BS + 4096)        /* f32 W_ih [1024]        = 4KB    */
#define OFF_XT    (OFF_WIH + 4096)       /* f32 x_t [16]                    */
#define SMEM_BYTES (OFF_XT + 64)         /* = 323,648 < 327,680 (320KB)     */

__device__ __forceinline__ unsigned char f32_to_e4m3(float f) {
  unsigned u = __float_as_uint(f);
  unsigned s = (u >> 24) & 0x80u;
  int e = (int)((u >> 23) & 0xFFu) - 127;
  unsigned man = u & 0x7FFFFFu;
  if (e < -6) return (unsigned char)s;              /* flush tiny/denorm */
  if (e > 8)  return (unsigned char)(s | 0x7Eu);    /* clamp to 448      */
  unsigned m3 = (man + 0x80000u) >> 20;             /* round mantissa->3b */
  if (m3 == 8u) { m3 = 0u; ++e; if (e > 8) return (unsigned char)(s | 0x7Eu); }
  if (e == 8 && m3 == 7u) m3 = 6u;                  /* avoid NaN 0x7F    */
  return (unsigned char)(s | ((unsigned)(e + 7) << 3) | m3);
}

__device__ __forceinline__ unsigned short f32_to_bf16(float f) {
  unsigned u = __float_as_uint(f);
  u += 0x7FFFu + ((u >> 16) & 1u);
  return (unsigned short)(u >> 16);
}
__device__ __forceinline__ float bf16_to_f32(unsigned short h) {
  return __uint_as_float(((unsigned)h) << 16);
}
__device__ __forceinline__ float sigmoidf_(float x) {
  return 1.0f / (1.0f + __expf(-x));
}
__device__ __forceinline__ float tanhf_(float x) {
  return 2.0f / (1.0f + __expf(-2.0f * x)) - 1.0f;
}

__global__ __launch_bounds__(256, 1)
void lstm_fused_fp8_wmma(const float* __restrict__ x,      /* [B,T,1]   */
                         const float* __restrict__ W_ih,   /* [1024,1]  */
                         const float* __restrict__ W_hh,   /* [1024,256]*/
                         const float* __restrict__ b_ih,   /* [1024]    */
                         const float* __restrict__ b_hh,   /* [1024]    */
                         const float* __restrict__ W_lin,  /* [512,256] */
                         const float* __restrict__ b_lin,  /* [512]     */
                         float* __restrict__ out)          /* [B,512]   */
{
  extern __shared__ char smem[];
  int*            Wq    = (int*)           (smem + OFF_W);     /* [1024][64] ints */
  unsigned short* gates = (unsigned short*)(smem + OFF_GATES); /* [16][1024] bf16 */
  int*            Hq    = (int*)           (smem + OFF_H8);    /* [16][64] ints   */
  float*          Hf    = (float*)         (smem + OFF_HF32);  /* [16][256]       */
  float*          BS    = (float*)         (smem + OFF_BS);
  float*          WI    = (float*)         (smem + OFF_WIH);
  float*          XT    = (float*)         (smem + OFF_XT);

  const int tid  = threadIdx.x;       /* 256 threads = 8 waves (wave32) */
  const int lane = tid & 31;
  const int wv   = tid >> 5;
  const int b0   = blockIdx.x * BTILE;
  const int half = lane >> 4;         /* 0: lanes 0-15, 1: lanes 16-31  */
  const int l15  = lane & 15;

  /* ---- init: W_hh f32 -> fp8 in LDS (read once, reused 512x) ---- */
  const float4* Wg4 = (const float4*)W_hh;
  for (int i = tid; i < 65536; i += 256) {
    float4 w = Wg4[i];
    unsigned p =  (unsigned)f32_to_e4m3(w.x)
               | ((unsigned)f32_to_e4m3(w.y) << 8)
               | ((unsigned)f32_to_e4m3(w.z) << 16)
               | ((unsigned)f32_to_e4m3(w.w) << 24);
    Wq[i] = (int)p;
  }
  for (int i = tid; i < 1024; i += 256) {
    BS[i] = b_ih[i] + b_hh[i];
    WI[i] = W_ih[i];
    Hq[i] = 0;                         /* h0 = 0 (fp8 zeros) */
  }
  __syncthreads();

  /* per-lane elementwise constants: lane `tid` owns h-column j = tid */
  const int j = tid;
  const float wi_i = WI[j],       wi_f = WI[256 + j],
              wi_g = WI[512 + j], wi_o = WI[768 + j];
  const float bs_i = BS[j],       bs_f = BS[256 + j],
              bs_g = BS[512 + j], bs_o = BS[768 + j];
  float c[BTILE];
  #pragma unroll
  for (int m = 0; m < BTILE; ++m) c[m] = 0.0f;

  const v8f vzero = {0.f, 0.f, 0.f, 0.f, 0.f, 0.f, 0.f, 0.f};

  for (int t = 0; t < T_LEN; ++t) {
    /* stage x_t for this tile (consumed after the barrier) */
    if (tid < BTILE) XT[tid] = x[(size_t)(b0 + tid) * T_LEN + t];

    /* ---- recurrent GEMM: gates[16][1024] = h[16][256] @ W_hh^T ----
       wave wv owns gate columns [128*wv, 128*wv+128): 8 N-tiles, 2 K-steps */
    v8f acc[8];
    #pragma unroll
    for (int nt = 0; nt < 8; ++nt) acc[nt] = vzero;

    #pragma unroll
    for (int ks = 0; ks < 2; ++ks) {
      const int kw = ks * 32 + half * 16;                  /* int offset in row */
      v16i a = *(const v16i*)(Hq + l15 * 64 + kw);         /* A: 16x128 fp8     */
      #pragma unroll
      for (int nt = 0; nt < 8; ++nt) {
        const int n = wv * 128 + nt * 16 + l15;            /* gate column       */
        v16i b = *(const v16i*)(Wq + n * 64 + kw);         /* B: 128x16 fp8     */
        acc[nt] = __builtin_amdgcn_wmma_f32_16x16x128_fp8_fp8(
            a, b, (short)0, acc[nt], false, false);
      }
    }

    /* stage accumulators to LDS (bf16) for the cross-wave gate combine */
    #pragma unroll
    for (int nt = 0; nt < 8; ++nt) {
      const int n = wv * 128 + nt * 16 + l15;
      #pragma unroll
      for (int r = 0; r < 8; ++r) {
        const int m = r + half * 8;                        /* C-layout row      */
        gates[m * G_DIM + n] = f32_to_bf16(acc[nt][r]);
      }
    }
    __syncthreads();

    /* ---- elementwise LSTM cell: lane owns h-column j, all 16 rows ---- */
    #pragma unroll 4
    for (int m = 0; m < BTILE; ++m) {
      const float xv = XT[m];
      float gi = bf16_to_f32(gates[m * G_DIM + j      ]) + xv * wi_i + bs_i;
      float gf = bf16_to_f32(gates[m * G_DIM + 256 + j]) + xv * wi_f + bs_f;
      float gg = bf16_to_f32(gates[m * G_DIM + 512 + j]) + xv * wi_g + bs_g;
      float go = bf16_to_f32(gates[m * G_DIM + 768 + j]) + xv * wi_o + bs_o;
      const float iv = sigmoidf_(gi);
      const float fv = sigmoidf_(gf);
      const float gv = tanhf_(gg);
      const float ov = sigmoidf_(go);
      c[m] = fv * c[m] + iv * gv;
      const float hn = ov * tanhf_(c[m]);
      ((unsigned char*)Hq)[m * H_DIM + j] = f32_to_e4m3(hn); /* feed next step */
      if (t == T_LEN - 1) Hf[m * H_DIM + j] = hn;            /* keep f32 last  */
    }
    __syncthreads();
  }

  /* ---- output head: out[16][512] = h_last @ W_lin^T + b_lin (L2-hot) ---- */
  for (int idx = tid; idx < BTILE * OUT_DIM; idx += 256) {
    const int m = idx >> 9;
    const int o = idx & (OUT_DIM - 1);
    const float4* wl = (const float4*)(W_lin + (size_t)o * H_DIM);
    const float4* hv = (const float4*)(Hf + m * H_DIM);
    float s = b_lin[o];
    #pragma unroll 8
    for (int k = 0; k < H_DIM / 4; ++k) {
      const float4 a = hv[k], w = wl[k];
      s += a.x * w.x + a.y * w.y + a.z * w.z + a.w * w.w;
    }
    out[(size_t)(b0 + m) * OUT_DIM + o] = s;
  }
}

extern "C" void kernel_launch(void* const* d_in, const int* in_sizes, int n_in,
                              void* d_out, int out_size, void* d_ws, size_t ws_size,
                              hipStream_t stream) {
  (void)in_sizes; (void)n_in; (void)out_size; (void)d_ws; (void)ws_size;
  const float* x     = (const float*)d_in[0];
  const float* W_ih  = (const float*)d_in[1];
  const float* W_hh  = (const float*)d_in[2];
  const float* b_ih  = (const float*)d_in[3];
  const float* b_hh  = (const float*)d_in[4];
  const float* W_lin = (const float*)d_in[5];
  const float* b_lin = (const float*)d_in[6];
  float* out = (float*)d_out;

  hipFuncSetAttribute((const void*)lstm_fused_fp8_wmma,
                      hipFuncAttributeMaxDynamicSharedMemorySize, SMEM_BYTES);

  lstm_fused_fp8_wmma<<<dim3(B_TOT / BTILE), dim3(256), SMEM_BYTES, stream>>>(
      x, W_ih, W_hh, b_ih, b_hh, W_lin, b_lin, out);
}
